// StreamingMultiScaleTCN_33792802684965
// MI455X (gfx1250) — compile-verified, matrix operand order checked by reading, not run on hardware
//
#include <hip/hip_runtime.h>
#include <hip/hip_bf16.h>

typedef __bf16 bf16_t;
typedef __attribute__((ext_vector_type(16))) __bf16 v16bf;
typedef __attribute__((ext_vector_type(8)))  __bf16 v8bf;
typedef __attribute__((ext_vector_type(4)))  __bf16 v4bf;
typedef __attribute__((ext_vector_type(8)))  float  v8f;

#define B_BATCH 32
#define T_LEN   4096
#define CIN     192
#define FDIM    320
#define HDIM    80
#define HIST    96           // max causal lookback: (9-1)*12
#define ROWS_PER_BLOCK 64
#define XROWS  (ROWS_PER_BLOCK + HIST)   // 160

// LDS layout (bytes)
#define LDS_X_BYTES   (XROWS * CIN * 2)            // 61440
#define LDS_XC_BYTES  (4 * 16 * FDIM * 2)          // 40960 (per-wave 16x320 bf16)
#define LDS_ATT_BYTES (4 * 16 * 96 * 2)            // 12288 (per-wave 16x96 bf16, K-padded)
#define LDS_TOTAL     (LDS_X_BYTES + LDS_XC_BYTES + LDS_ATT_BYTES)

// ---------------------------------------------------------------------------
// Weight repack: row-major [K,N] f32 -> WMMA bf16 B-fragments (32x16 per frag).
// Fragment layout per lane (mirror of 16-bit A layout):
//   lane<16 : N=lane,    K = {0..7, 16..23} + kc*32
//   lane>=16: N=lane-16, K = {8..15,24..31} + kc*32
// Each lane stores 16 contiguous bf16 (32B) at frag*512 + lane*16.
// ---------------------------------------------------------------------------
__global__ void prep_frags(const float* __restrict__ src, int K, int N,
                           bf16_t* __restrict__ dst) {
    int frag  = blockIdx.x;
    int ntiles = N >> 4;
    int kc = frag / ntiles;
    int nt = frag % ntiles;
    int lane = threadIdx.x;          // 0..31
    int n = nt * 16 + (lane & 15);
    int hi = (lane >= 16) ? 8 : 0;
    bf16_t vals[16];
#pragma unroll
    for (int e = 0; e < 16; ++e) {
        int vg = e >> 1, half = e & 1;
        int k = kc * 32 + ((vg < 4) ? 0 : 16) + hi + ((vg & 3) * 2) + half;
        float v = (k < K) ? src[(size_t)k * N + n] : 0.0f;
        vals[e] = (bf16_t)v;
    }
    bf16_t* p = dst + (size_t)frag * 512 + lane * 16;
#pragma unroll
    for (int e = 0; e < 16; ++e) p[e] = vals[e];
}

// ---------------------------------------------------------------------------
// Device helpers
// ---------------------------------------------------------------------------
__device__ __forceinline__ v8f wmma_bf16(v16bf a, v16bf b, v8f c) {
    return __builtin_amdgcn_wmma_f32_16x16x32_bf16(
        false, a, false, b, (short)0, c, false, false);
}

// A fragment (16x32 bf16) from row-major bf16 LDS tile: two ds_load_b128.
__device__ __forceinline__ v16bf load_a_frag(const bf16_t* tile, int rowStride,
                                             int row0, int kc, int laneHi, int laneLo) {
    const char* base = (const char*)tile;
    int byteOff = (row0 + laneLo) * rowStride * 2 + kc * 64 + (laneHi ? 16 : 0);
    v8bf lo = *(const v8bf*)(base + byteOff);
    v8bf hi = *(const v8bf*)(base + byteOff + 32);
    v16bf a;
#pragma unroll
    for (int i = 0; i < 8; ++i) { a[i] = lo[i]; a[i + 8] = hi[i]; }
    return a;
}

// B fragment: 32 contiguous bytes per lane from repacked workspace.
__device__ __forceinline__ v16bf load_b_frag(const bf16_t* frags, int fragIdx, int lane) {
    return *(const v16bf*)(frags + (size_t)fragIdx * 512 + lane * 16);
}

// In-order DS pipe per wave; this stops compiler reordering + drains DScnt so
// cross-lane LDS reads after same-wave LDS writes are safe without a barrier.
__device__ __forceinline__ void wave_lds_fence() {
    asm volatile("s_wait_dscnt 0" ::: "memory");
}

// ---------------------------------------------------------------------------
// Fused TCN kernel: 128 threads = 4 waves; wave w owns rows t0_blk+16w .. +15.
// N-register-blocked: one A fragment feeds 4-5 independent WMMA accumulators.
// ---------------------------------------------------------------------------
__global__ void __launch_bounds__(128)
tcn_fused(const float* __restrict__ x,
          const bf16_t* __restrict__ wConv,   // [29 taps][6 kc][4 nt][512]
          const bf16_t* __restrict__ wA1,     // [10 kc][5 nt][512]
          const bf16_t* __restrict__ wA2,     // [3 kc][20 nt][512]
          const bf16_t* __restrict__ wO,      // [10 kc][20 nt][512]
          const float* __restrict__ bb1, const float* __restrict__ bb2,
          const float* __restrict__ bb3, const float* __restrict__ bb4,
          const float* __restrict__ bb5,
          const float* __restrict__ bA1, const float* __restrict__ bA2,
          const float* __restrict__ bOut,
          float* __restrict__ y) {
    extern __shared__ char smem[];
    bf16_t* xTile = (bf16_t*)smem;                                   // [160][192]
    const int lane   = threadIdx.x & 31;
    const int wave   = threadIdx.x >> 5;
    const int laneLo = lane & 15;
    const int laneHi = (lane >= 16) ? 1 : 0;
    bf16_t* xcW  = (bf16_t*)(smem + LDS_X_BYTES) + wave * 16 * FDIM; // [16][320]
    bf16_t* attW = (bf16_t*)(smem + LDS_X_BYTES + LDS_XC_BYTES) + wave * 16 * 96;

    const int bidx   = blockIdx.x >> 6;   // 64 blocks per batch (T/64)
    const int t0_blk = (blockIdx.x & 63) * ROWS_PER_BLOCK;
    const float* xb = x + (size_t)bidx * T_LEN * CIN;

    // ---- Phase 1: cooperative stage of x[t0_blk-96 .. t0_blk+63] -> LDS bf16
    for (int i = threadIdx.x; i < XROWS * (CIN / 4); i += 128) {
        int row = i / (CIN / 4);
        int c4  = i % (CIN / 4);
        int gt  = t0_blk - HIST + row;
        float4 v = make_float4(0.f, 0.f, 0.f, 0.f);
        if (gt >= 0) v = *(const float4*)(xb + (size_t)gt * CIN + c4 * 4);
        v4bf o; o[0] = (bf16_t)v.x; o[1] = (bf16_t)v.y; o[2] = (bf16_t)v.z; o[3] = (bf16_t)v.w;
        *(v4bf*)(xTile + row * CIN + c4 * 4) = o;
    }
    __syncthreads();

    // ---- Phase 2: 5 dilated causal conv branches. One A fragment feeds the
    //      branch's 4 output tiles (4 independent accumulator chains).
    const int KS[5]      = {3, 3, 5, 9, 9};
    const int DIL[5]     = {1, 2, 4, 8, 12};
    const int TAPBASE[5] = {0, 3, 6, 11, 20};
    const float* BPTR[5] = {bb1, bb2, bb3, bb4, bb5};
    const int rowBase = wave * 16 + HIST;   // LDS row of this wave's t0

    for (int br = 0; br < 5; ++br) {
        v8f acc[4];
#pragma unroll
        for (int nt = 0; nt < 4; ++nt) acc[nt] = (v8f){};
        for (int j = 0; j < KS[br]; ++j) {
            int row0 = rowBase - (KS[br] - 1 - j) * DIL[br];
            int tap  = TAPBASE[br] + j;
#pragma unroll
            for (int kc = 0; kc < 6; ++kc) {
                v16bf a = load_a_frag(xTile, CIN, row0, kc, laneHi, laneLo);
#pragma unroll
                for (int nt = 0; nt < 4; ++nt) {
                    v16bf bm = load_b_frag(wConv, (tap * 6 + kc) * 4 + nt, lane);
                    acc[nt] = wmma_bf16(a, bm, acc[nt]);
                }
            }
        }
#pragma unroll
        for (int nt = 0; nt < 4; ++nt) {
            float bias = BPTR[br][nt * 16 + laneLo];
            int colBase = (br * 4 + nt) * 16;
#pragma unroll
            for (int r = 0; r < 8; ++r) {
                float v = fmaxf(acc[nt][r] + bias, 0.0f);
                int row = r + laneHi * 8;
                xcW[row * FDIM + colBase + laneLo] = (bf16_t)v;
            }
        }
    }
    wave_lds_fence();

    // ---- Phase 3: att1 = relu(xc[16x320] @ wA1[320x80] + b). 5 accumulators.
    {
        v8f acc[5];
#pragma unroll
        for (int nt = 0; nt < 5; ++nt) acc[nt] = (v8f){};
#pragma unroll
        for (int kc = 0; kc < 10; ++kc) {
            v16bf a = load_a_frag(xcW, FDIM, 0, kc, laneHi, laneLo);
#pragma unroll
            for (int nt = 0; nt < 5; ++nt) {
                v16bf bm = load_b_frag(wA1, kc * 5 + nt, lane);
                acc[nt] = wmma_bf16(a, bm, acc[nt]);
            }
        }
#pragma unroll
        for (int nt = 0; nt < 5; ++nt) {
            float bias = bA1[nt * 16 + laneLo];
#pragma unroll
            for (int r = 0; r < 8; ++r) {
                float v = fmaxf(acc[nt][r] + bias, 0.0f);
                int row = r + laneHi * 8;
                attW[row * 96 + nt * 16 + laneLo] = (bf16_t)v;
            }
        }
    }
    // zero K-pad columns 80..95 (wA2 was packed with K padded 80->96)
#pragma unroll
    for (int r = 0; r < 8; ++r)
        attW[(r + laneHi * 8) * 96 + 80 + laneLo] = (bf16_t)0.0f;
    wave_lds_fence();

    // ---- Phase 4: att2 = sigmoid(att1[16x96] @ wA2[96x320] + b); gate xc.
    //      20 N-tiles in 4 groups of 5 accumulators.
    for (int g = 0; g < 4; ++g) {
        v8f acc[5];
#pragma unroll
        for (int i = 0; i < 5; ++i) acc[i] = (v8f){};
#pragma unroll
        for (int kc = 0; kc < 3; ++kc) {
            v16bf a = load_a_frag(attW, 96, 0, kc, laneHi, laneLo);
#pragma unroll
            for (int i = 0; i < 5; ++i) {
                v16bf bm = load_b_frag(wA2, kc * 20 + g * 5 + i, lane);
                acc[i] = wmma_bf16(a, bm, acc[i]);
            }
        }
#pragma unroll
        for (int i = 0; i < 5; ++i) {
            int nt = g * 5 + i;
            float bias = bA2[nt * 16 + laneLo];
#pragma unroll
            for (int r = 0; r < 8; ++r) {
                float s = 1.0f / (1.0f + __expf(-(acc[i][r] + bias)));
                int row = r + laneHi * 8;
                int idx = row * FDIM + nt * 16 + laneLo;
                float xcv = (float)xcW[idx];
                xcW[idx] = (bf16_t)(xcv * s);   // xa overwrites xc (used once)
            }
        }
    }
    wave_lds_fence();

    // ---- Phase 5: y = xa[16x320] @ wOut[320x320] + b_out  (f32 out).
    //      20 N-tiles in 4 groups of 5 accumulators; A reused 5x per load.
    const int t0 = t0_blk + wave * 16;
    float* yb = y + ((size_t)bidx * T_LEN + t0) * FDIM;
    for (int g = 0; g < 4; ++g) {
        v8f acc[5];
#pragma unroll
        for (int i = 0; i < 5; ++i) acc[i] = (v8f){};
#pragma unroll
        for (int kc = 0; kc < 10; ++kc) {
            v16bf a = load_a_frag(xcW, FDIM, 0, kc, laneHi, laneLo);
#pragma unroll
            for (int i = 0; i < 5; ++i) {
                v16bf bm = load_b_frag(wO, kc * 20 + g * 5 + i, lane);
                acc[i] = wmma_bf16(a, bm, acc[i]);
            }
        }
#pragma unroll
        for (int i = 0; i < 5; ++i) {
            int nt = g * 5 + i;
            float bias = bOut[nt * 16 + laneLo];
#pragma unroll
            for (int r = 0; r < 8; ++r) {
                int row = r + laneHi * 8;
                yb[(size_t)row * FDIM + nt * 16 + laneLo] = acc[i][r] + bias;
            }
        }
    }
}

// ---------------------------------------------------------------------------
// Host launcher
// ---------------------------------------------------------------------------
extern "C" void kernel_launch(void* const* d_in, const int* in_sizes, int n_in,
                              void* d_out, int out_size, void* d_ws, size_t ws_size,
                              hipStream_t stream) {
    const float* x      = (const float*)d_in[0];
    const float* w_b[5] = {(const float*)d_in[1], (const float*)d_in[3],
                           (const float*)d_in[5], (const float*)d_in[7],
                           (const float*)d_in[9]};
    const float* b_b[5] = {(const float*)d_in[2], (const float*)d_in[4],
                           (const float*)d_in[6], (const float*)d_in[8],
                           (const float*)d_in[10]};
    const float* w_att1 = (const float*)d_in[11];
    const float* b_att1 = (const float*)d_in[12];
    const float* w_att2 = (const float*)d_in[13];
    const float* b_att2 = (const float*)d_in[14];
    const float* w_out  = (const float*)d_in[15];
    const float* b_out  = (const float*)d_in[16];
    float* out = (float*)d_out;

    // Workspace carving (bf16 elements). Total ~1.0 MB.
    bf16_t* wsConv = (bf16_t*)d_ws;                 // 29*6*4*512
    bf16_t* wsA1   = wsConv + 29 * 6 * 4 * 512;     // 10*5*512
    bf16_t* wsA2   = wsA1 + 10 * 5 * 512;           // 3*20*512 (K padded)
    bf16_t* wsO    = wsA2 + 3 * 20 * 512;           // 10*20*512

    // Repack weights into WMMA B-fragment order (cheap; graph-capture safe).
    const int ks[5] = {3, 3, 5, 9, 9};
    bf16_t* dst = wsConv;
    for (int br = 0; br < 5; ++br) {
        for (int j = 0; j < ks[br]; ++j) {
            prep_frags<<<6 * 4, 32, 0, stream>>>(w_b[br] + (size_t)j * CIN * 64,
                                                 CIN, 64, dst);
            dst += 6 * 4 * 512;
        }
    }
    prep_frags<<<10 * 5, 32, 0, stream>>>(w_att1, FDIM, HDIM, wsA1);
    prep_frags<<<3 * 20, 32, 0, stream>>>(w_att2, HDIM, FDIM, wsA2); // K 80 -> pad 96
    prep_frags<<<10 * 20, 32, 0, stream>>>(w_out, FDIM, FDIM, wsO);

    // Fused main kernel: 2048 blocks x 128 threads (4 waves), 112KB dynamic LDS.
    tcn_fused<<<B_BATCH * (T_LEN / ROWS_PER_BLOCK), 128, LDS_TOTAL, stream>>>(
        x, wsConv, wsA1, wsA2, wsO,
        b_b[0], b_b[1], b_b[2], b_b[3], b_b[4],
        b_att1, b_att2, b_out, out);
}
